// VQModule_36636071035462
// MI455X (gfx1250) — compile-verified, matrix operand order checked by reading, not run on hardware
//
#include <hip/hip_runtime.h>
#include <hip/hip_bf16.h>

typedef __bf16 bf16_t;
typedef __bf16 v16bf __attribute__((ext_vector_type(16)));
typedef __bf16 v8bf  __attribute__((ext_vector_type(8)));
typedef float  v8f   __attribute__((ext_vector_type(8)));
typedef int    v4i   __attribute__((ext_vector_type(4)));

#if defined(__has_builtin)
#if __has_builtin(__builtin_amdgcn_global_load_async_to_lds_b128)
#define HAVE_ASYNC 1
#endif
#endif

typedef __attribute__((address_space(1))) v4i gv4i;   // global
typedef __attribute__((address_space(3))) v4i lv4i;   // LDS

#define D_CODES 2048
#define C_DIM   256
#define N_ROWS  65536
#define ROWS_PER_WG 128
#define CODES_PER_STAGE 32
#define N_STAGES (D_CODES / CODES_PER_STAGE)          // 64
#define BBUF_ELEMS (CODES_PER_STAGE * C_DIM)          // 8192 bf16 = 16KB
#define MOM     0.99f
#define EPSV    1e-5f

// ---- output offsets (floats): quantized, loss, ids, new_embed, new_cs, new_ea
#define OFF_Q     0u
#define OFF_LOSS  16777216u
#define OFF_IDS   16777217u
#define OFF_NEMB  16842753u
#define OFF_NCS   17367041u
#define OFF_NEA   17369089u

// ---- workspace offsets (floats)
#define WS_ONEHOT 0
#define WS_ESUM   2048
#define WS_SCAL   526336        // [0]=loss sum, [1]=sum(new_cluster_size)
#define WS_NORM   526344
#define WS_EBF    528392        // bf16 codebook (2048*256 bf16)
#define WS_ZERO_N 526344

union Frag16 { v16bf v; v8bf h[2]; };

__global__ void vq_zero(float* __restrict__ ws, int n) {
    int i = blockIdx.x * 256 + threadIdx.x;
    if (i < n) ws[i] = 0.0f;
}

__global__ void vq_prep(const float* __restrict__ embed,
                        bf16_t* __restrict__ ebf,
                        float* __restrict__ normsq) {
    int t = threadIdx.x;
    int code = blockIdx.x * 8 + (t >> 5);
    int lane = t & 31;
    const float* row = embed + (size_t)code * C_DIM;
    float ss = 0.0f;
#pragma unroll
    for (int i = 0; i < 8; ++i) {
        int k = lane * 8 + i;
        float e = row[k];
        ebf[(size_t)code * C_DIM + k] = (bf16_t)e;
        ss += e * e;
    }
#pragma unroll
    for (int off = 16; off >= 1; off >>= 1) ss += __shfl_xor(ss, off, 32);
    if (lane == 0) normsq[code] = ss;
}

// Stage one 32-code x 256-dim bf16 tile (16KB) into LDS; 64B per thread.
__device__ __forceinline__ void stage_b(const bf16_t* __restrict__ ebf,
                                        bf16_t* dst, int s, int t) {
    const char* src = (const char*)(ebf + (size_t)s * BBUF_ELEMS) + t * 64;
    char* d = (char*)dst + t * 64;
#ifdef HAVE_ASYNC
    gv4i* g = (gv4i*)(uintptr_t)src;
    lv4i* l = (lv4i*)(unsigned)(uintptr_t)d;
    __builtin_amdgcn_global_load_async_to_lds_b128(g + 0, l + 0, 0, 0);
    __builtin_amdgcn_global_load_async_to_lds_b128(g + 1, l + 1, 0, 0);
    __builtin_amdgcn_global_load_async_to_lds_b128(g + 2, l + 2, 0, 0);
    __builtin_amdgcn_global_load_async_to_lds_b128(g + 3, l + 3, 0, 0);
#else
    const v4i* sg = (const v4i*)src;
    v4i* dl = (v4i*)d;
    dl[0] = sg[0]; dl[1] = sg[1]; dl[2] = sg[2]; dl[3] = sg[3];
#endif
}

// Main fused kernel: 512 WGs x 256 threads (8 waves x 16 rows).
__global__ void __launch_bounds__(256)
vq_main(const float* __restrict__ x, const float* __restrict__ embed,
        const bf16_t* __restrict__ ebf, const float* __restrict__ normsq,
        float* __restrict__ onehot, float* __restrict__ esum,
        float* __restrict__ scal, float* __restrict__ out) {
    // 64KB region, reused: phase 1 = x staging (128x256 bf16);
    // phase 2 = B double buffer (2 x 16KB).
    __shared__ __align__(16) bf16_t ldsS[ROWS_PER_WG * C_DIM];

    const int t = threadIdx.x;
    const int rowBase = blockIdx.x * ROWS_PER_WG;

    // ---- phase 1: stage 128 rows of flat(x) into LDS as bf16 ----
    {
        int r  = t >> 1;
        int ks = (t & 1) * 128;
        int n  = rowBase + r;
        int bi = n >> 12, wi = (n >> 6) & 63, hi = n & 63;
        const float* xp = x + (size_t)bi * 1048576 + (size_t)hi * 64 + wi;
#pragma unroll 4
        for (int i = 0; i < 128; ++i) {
            int k = ks + i;
            ldsS[r * C_DIM + k] = (bf16_t)xp[(size_t)k * 4096];
        }
    }
    __syncthreads();

    const int wave    = t >> 5;
    const int lane    = t & 31;
    const int half    = lane >> 4;
    const int m       = lane & 15;
    const int waveRow = wave * 16;

    // ---- load this wave's 8 A fragments (16 rows x K=256) from LDS ----
    Frag16 a[8];
    {
        const bf16_t* ap = &ldsS[(waveRow + m) * C_DIM];
#pragma unroll
        for (int kk = 0; kk < 8; ++kk) {
            int kb = kk * 32 + half * 8;
            a[kk].h[0] = *(const v8bf*)(ap + kb);
            a[kk].h[1] = *(const v8bf*)(ap + kb + 16);
        }
    }
    // Our DS reads must land before anyone overwrites the region with B tiles.
    asm volatile("s_wait_dscnt 0x0" ::: "memory");
    __syncthreads();

    float best[8];
    int   bidx[8];
#pragma unroll
    for (int v = 0; v < 8; ++v) { best[v] = -3.4e38f; bidx[v] = 0; }

    // ---- phase 2: async double-buffered codebook sweep ----
    stage_b(ebf, ldsS, 0, t);
    for (int s = 0; s < N_STAGES; ++s) {
        asm volatile("s_wait_asynccnt 0x0" ::: "memory");
        __syncthreads();                       // staged tile visible WG-wide
        if (s + 1 < N_STAGES) {
            stage_b(ebf, ldsS + ((s + 1) & 1) * BBUF_ELEMS, s + 1, t);
            __builtin_prefetch(ebf + (size_t)(s + 2) * BBUF_ELEMS, 0, 0);
        }
        const bf16_t* bt = ldsS + (s & 1) * BBUF_ELEMS;
#pragma unroll
        for (int sub = 0; sub < 2; ++sub) {
            const int tile = s * 2 + sub;
            const int code = tile * 16 + m;
            const float ns = normsq[code];
            const bf16_t* bp = bt + (sub * 16 + m) * C_DIM + half * 8;
            v8f c = {};
#pragma unroll
            for (int kk = 0; kk < 8; ++kk) {
                Frag16 b;
                b.h[0] = *(const v8bf*)(bp + kk * 32);
                b.h[1] = *(const v8bf*)(bp + kk * 32 + 16);
                c = __builtin_amdgcn_wmma_f32_16x16x32_bf16(
                        false, a[kk].v, false, b.v, (short)0, c, false, false);
            }
#pragma unroll
            for (int v = 0; v < 8; ++v) {
                float sc = 2.0f * c[v] - ns;
                if (sc > best[v]) { best[v] = sc; bidx[v] = code; }
            }
        }
        __syncthreads();                       // reads done before buffer reuse
    }

    // ---- argmax reduce across each 16-lane half, then fused writeback ----
    float lsum = 0.0f;
#pragma unroll
    for (int v = 0; v < 8; ++v) {
        float bv = best[v];
        int   bc = bidx[v];
#pragma unroll
        for (int off = 8; off >= 1; off >>= 1) {
            float ov = __shfl_xor(bv, off, 32);
            int   oc = __shfl_xor(bc, off, 32);
            if (ov > bv || (ov == bv && oc < bc)) { bv = ov; bc = oc; }
        }
        int r = waveRow + half * 8 + v;
        int n = rowBase + r;
        int bi = n >> 12, wi = (n >> 6) & 63, hi = n & 63;
        if (m == 0) {
            out[OFF_IDS + n] = (float)bc;
            atomicAdd(&onehot[bc], 1.0f);
        }
        const float* xrow = x   + (size_t)bi * 1048576 + (size_t)hi * 64 + wi;
        float*       qout = out + OFF_Q + (size_t)bi * 1048576 + (size_t)hi * 64 + wi;
        const float* er   = embed + (size_t)bc * C_DIM;
#pragma unroll 4
        for (int kb = 0; kb < 16; ++kb) {
            int k = m * 16 + kb;
            float q  = er[k];
            float xv = xrow[(size_t)k * 4096];
            qout[(size_t)k * 4096] = q;
            float d = xv - q;
            lsum += d * d;
            atomicAdd(&esum[(size_t)k * D_CODES + bc], xv);
        }
    }
    atomicAdd(&scal[0], lsum);
}

__global__ void vq_fin1(const float* __restrict__ cluster_size,
                        const float* __restrict__ onehot,
                        float* __restrict__ scal, float* __restrict__ out) {
    __shared__ float red[256];
    int t = threadIdx.x;
    float p = 0.0f;
    for (int i = t; i < D_CODES; i += 256) {
        float ncs = MOM * cluster_size[i] + (1.0f - MOM) * onehot[i];
        out[OFF_NCS + i] = ncs;
        p += ncs;
    }
    red[t] = p;
    __syncthreads();
    for (int s = 128; s > 0; s >>= 1) {
        if (t < s) red[t] += red[t + s];
        __syncthreads();
    }
    if (t == 0) {
        scal[1] = red[0];
        out[OFF_LOSS] = scal[0] / (float)(16ull * 256ull * 64ull * 64ull);
    }
}

__global__ void vq_fin2(const float* __restrict__ embed_avg,
                        const float* __restrict__ esum,
                        const float* __restrict__ scal,
                        float* __restrict__ out) {
    int idx = blockIdx.x * 256 + threadIdx.x;
    int ci = idx >> 11;
    int d  = idx & 2047;
    float ea = MOM * embed_avg[idx] + (1.0f - MOM) * esum[idx];
    out[OFF_NEA + idx] = ea;
    float ncs = out[OFF_NCS + d];
    float nn  = scal[1];
    float cs  = nn * (ncs + EPSV) / (nn + (float)D_CODES * EPSV);
    out[OFF_NEMB + (size_t)d * C_DIM + ci] = ea / cs;
}

extern "C" void kernel_launch(void* const* d_in, const int* in_sizes, int n_in,
                              void* d_out, int out_size, void* d_ws, size_t ws_size,
                              hipStream_t stream) {
    const float* x     = (const float*)d_in[0];
    const float* embed = (const float*)d_in[1];
    const float* cs    = (const float*)d_in[2];
    const float* ea    = (const float*)d_in[3];
    float* out = (float*)d_out;
    float* ws  = (float*)d_ws;

    float*  onehot = ws + WS_ONEHOT;
    float*  esum   = ws + WS_ESUM;
    float*  scal   = ws + WS_SCAL;
    float*  normsq = ws + WS_NORM;
    bf16_t* ebf    = (bf16_t*)(ws + WS_EBF);

    vq_zero<<<(WS_ZERO_N + 255) / 256, 256, 0, stream>>>(ws, WS_ZERO_N);
    vq_prep<<<D_CODES / 8, 256, 0, stream>>>(embed, ebf, normsq);
    vq_main<<<N_ROWS / ROWS_PER_WG, 256, 0, stream>>>(x, embed, ebf, normsq,
                                                      onehot, esum, scal, out);
    vq_fin1<<<1, 256, 0, stream>>>(cs, onehot, scal, out);
    vq_fin2<<<(C_DIM * D_CODES) / 256, 256, 0, stream>>>(ea, esum, scal, out);
}
